// AttentionEncoder_86466281603703
// MI455X (gfx1250) — compile-verified, matrix operand order checked by reading, not run on hardware
//
#include <hip/hip_runtime.h>
#include <hip/hip_bf16.h>
#include <math.h>

// ---------------------------------------------------------------------------
// Types for CDNA5 WMMA (wave32): 16x16x32 bf16 -> f32
// ---------------------------------------------------------------------------
typedef __bf16 bf16;
typedef __attribute__((ext_vector_type(8)))  bf16  bf16x8;
typedef __attribute__((ext_vector_type(16))) bf16  bf16x16;
typedef __attribute__((ext_vector_type(8)))  float f32x8;

#define N_LAYER 6
#define N_HEAD  16
#define HIDDEN  1024
#define HEADD   64
#define FFWD    2048
#define SEQ     512
#define BATCH   16
#define MROWS   (BATCH * SEQ)       // 8192 token rows
#define RMS_EPS 1.1920929e-07f      // finfo(float32).eps

// ---------------------------------------------------------------------------
// Fragment load helpers.
// A-matrix 16x32 bf16 layout (ISA 7.12.2): lane holds row M=(lane&15);
// elems 0..7  -> K = k0 + (lane>>4)*8 + i      (contiguous 16B)
// elems 8..15 -> K = k0 + 16 + (lane>>4)*8 + i (contiguous 16B, +16 elems)
// `base` must already point at row_ptr + k0 + (lane>>4)*8.
// The B operand of q@k^T (i.e. k^T from row-major k) uses the same pattern
// with "row" = key column, so this helper serves both operands.
// ---------------------------------------------------------------------------
__device__ __forceinline__ bf16x16 load_frag_rowmajor(const bf16* base) {
    const bf16x8* p = (const bf16x8*)base;
    bf16x8 lo = p[0];        // elements [0..7]
    bf16x8 hi = p[2];        // +16 bf16 elements -> [8..15]
    bf16x16 r;
#pragma unroll
    for (int i = 0; i < 8; ++i) { r[i] = lo[i]; r[i + 8] = hi[i]; }
    return r;
}

// Pre-packed B fragments: each lane's 16 values are contiguous (32 bytes).
__device__ __forceinline__ bf16x16 load_frag_packed(const bf16* base) {
    const bf16x8* p = (const bf16x8*)base;
    bf16x8 lo = p[0], hi = p[1];
    bf16x16 r;
#pragma unroll
    for (int i = 0; i < 8; ++i) { r[i] = lo[i]; r[i + 8] = hi[i]; }
    return r;
}

// ---------------------------------------------------------------------------
// Weight pack: f32 [K][N] (or head-blocked [N/64][K][64]) -> bf16 in exact
// B-fragment order so the GEMM loads B with two global_load_b128 per lane.
// Linear index t = (((nt*(K/32) + ks)*32 + lane)*16 + i)
// ---------------------------------------------------------------------------
__global__ void pack_weight(const float* __restrict__ W, bf16* __restrict__ Wp,
                            int K, int N, int headed) {
    size_t t = (size_t)blockIdx.x * blockDim.x + threadIdx.x;
    size_t total = (size_t)K * (size_t)N;
    if (t >= total) return;
    int i    = (int)(t & 15);
    size_t u = t >> 4;
    int lane = (int)(u & 31);  u >>= 5;
    int ksteps = K >> 5;
    int ks = (int)(u % ksteps);
    int nt = (int)(u / ksteps);
    int half = lane >> 4;
    int k = ks * 32 + (i < 8 ? half * 8 + i : 16 + half * 8 + (i - 8));
    int n = nt * 16 + (lane & 15);
    float v;
    if (headed) {  // src = [N/64 heads][K][64]
        int hh = n >> 6, e = n & 63;
        v = W[(size_t)hh * K * 64 + (size_t)k * 64 + e];
    } else {       // src = [K][N]
        v = W[(size_t)k * N + n];
    }
    Wp[t] = (bf16)v;
}

// ---------------------------------------------------------------------------
// Embedding: h[b,l,:] = concat(emb[act], dur) + pos[l]
// ---------------------------------------------------------------------------
__global__ void embed_kernel(const float* __restrict__ x,
                             const float* __restrict__ emb,
                             const float* __restrict__ pos,
                             float* __restrict__ h) {
    int row = blockIdx.x;           // b*512 + l
    int l   = row & (SEQ - 1);
    int act = (int)x[(size_t)row * 2 + 0];
    float dur = x[(size_t)row * 2 + 1];
    for (int c = threadIdx.x; c < HIDDEN; c += blockDim.x) {
        float e = (c < HIDDEN - 1) ? emb[(size_t)act * (HIDDEN - 1) + c] : dur;
        h[(size_t)row * HIDDEN + c] = e + pos[(size_t)l * HIDDEN + c];
    }
}

// ---------------------------------------------------------------------------
// RMSNorm: out_bf16[row] = h[row] * rsqrt(mean(h^2)+eps) * g
// ---------------------------------------------------------------------------
__global__ __launch_bounds__(256)
void rmsnorm_kernel(const float* __restrict__ h, const float* __restrict__ g,
                    bf16* __restrict__ out) {
    __shared__ float red[8];
    __shared__ float invs;
    int row = blockIdx.x;
    const float* hr = h + (size_t)row * HIDDEN;
    float s = 0.f;
    for (int c = threadIdx.x; c < HIDDEN; c += 256) { float v = hr[c]; s += v * v; }
#pragma unroll
    for (int off = 16; off > 0; off >>= 1) s += __shfl_xor(s, off, 32);
    if ((threadIdx.x & 31) == 0) red[threadIdx.x >> 5] = s;
    __syncthreads();
    if (threadIdx.x == 0) {
        float t = 0.f;
#pragma unroll
        for (int i = 0; i < 8; ++i) t += red[i];
        invs = rsqrtf(t * (1.0f / HIDDEN) + RMS_EPS);
    }
    __syncthreads();
    float inv = invs;
    for (int c = threadIdx.x; c < HIDDEN; c += 256)
        out[(size_t)row * HIDDEN + c] = (bf16)(hr[c] * inv * g[c]);
}

// ---------------------------------------------------------------------------
// Register-blocked WMMA GEMM: C[M,N] = A[M,K] (bf16 row-major) @ Bp (packed).
// Each wave computes a 64x32 output block: 4 M-subtiles x 2 N-subtiles,
// 8 f32x8 accumulators. Per k-step: 4 A-frag loads + 2 B-frag loads
// (12 x global_load_b128) feed 8 WMMAs -> ~21 FLOP/byte from L2.
// MODE 0: outb = bf16(acc)                       (QKV projections)
// MODE 1: outf = resid + acc + bias              (Wo / FFN2 + residual)
// MODE 2: outb = bf16(gelu_exact(acc + bias))    (FFN1)
// ---------------------------------------------------------------------------
template <int MODE>
__global__ __launch_bounds__(256)
void gemm_bf16_wmma(const bf16* __restrict__ A, const bf16* __restrict__ Bp,
                    const float* __restrict__ bias,
                    const float* __restrict__ resid,
                    bf16* __restrict__ outb, float* __restrict__ outf,
                    int M, int N, int K) {
    int lane = threadIdx.x & 31;
    int wave = blockIdx.x * (blockDim.x >> 5) + (threadIdx.x >> 5);
    int tilesN = N >> 5;                 // 32-wide N blocks
    int mt = wave / tilesN;              // 64-row M block
    int nt = wave - mt * tilesN;
    if (mt >= (M >> 6)) return;
    int half = lane >> 4, lr = lane & 15;
    int ksteps = K >> 5;

    f32x8 acc[4][2];
#pragma unroll
    for (int s = 0; s < 4; ++s) {
        acc[s][0] = (f32x8){};
        acc[s][1] = (f32x8){};
    }

    const bf16* arow[4];
#pragma unroll
    for (int s = 0; s < 4; ++s)
        arow[s] = A + (size_t)(mt * 64 + s * 16 + lr) * K + half * 8;
    const bf16* bbase0 = Bp + (size_t)(nt * 2) * ksteps * 512 + (size_t)lane * 16;
    const bf16* bbase1 = bbase0 + (size_t)ksteps * 512;

    for (int ks = 0; ks < ksteps; ++ks) {
        bf16x16 b0 = load_frag_packed(bbase0 + (size_t)ks * 512);
        bf16x16 b1 = load_frag_packed(bbase1 + (size_t)ks * 512);
#pragma unroll
        for (int s = 0; s < 4; ++s) {
            __builtin_prefetch(arow[s] + (ks + 1) * 32, 0, 0);
            bf16x16 a = load_frag_rowmajor(arow[s] + ks * 32);
            acc[s][0] = __builtin_amdgcn_wmma_f32_16x16x32_bf16(
                false, a, false, b0, (short)0, acc[s][0], false, false);
            acc[s][1] = __builtin_amdgcn_wmma_f32_16x16x32_bf16(
                false, a, false, b1, (short)0, acc[s][1], false, false);
        }
    }

#pragma unroll
    for (int j = 0; j < 2; ++j) {
        int n = nt * 32 + j * 16 + lr;
        float bv = (MODE != 0) ? bias[n] : 0.f;
#pragma unroll
        for (int s = 0; s < 4; ++s) {
#pragma unroll
            for (int r = 0; r < 8; ++r) {
                int m = mt * 64 + s * 16 + half * 8 + r;
                size_t idx = (size_t)m * N + n;
                if (MODE == 0) {
                    outb[idx] = (bf16)acc[s][j][r];
                } else if (MODE == 1) {
                    outf[idx] = resid[idx] + acc[s][j][r] + bv;
                } else {  // MODE == 2 : exact GELU
                    float t = acc[s][j][r] + bv;
                    outb[idx] = (bf16)(0.5f * t *
                                       (1.0f + erff(t * 0.70710678118654752f)));
                }
            }
        }
    }
}

// ---------------------------------------------------------------------------
// Attention: one wave per (b, head, 16-query tile).
//  scores (16x512) = q @ k^T * scale  via WMMA  -> LDS (f32)
//  softmax (wave32 shuffle-combined halves)     -> LDS (bf16 probs)
//  att (16x64) = P @ V via WMMA (V staged through LDS)
// Output written as [b*512+l][h*64+e] so Wo GEMM is a standard GEMM.
// ---------------------------------------------------------------------------
__global__ __launch_bounds__(32)
void attention_kernel(const bf16* __restrict__ Q, const bf16* __restrict__ Kb,
                      const bf16* __restrict__ Vb, bf16* __restrict__ Ab) {
    __shared__ float S[16 * SEQ];       // 32 KB scores
    __shared__ bf16  P[16 * SEQ];       // 16 KB probs
    __shared__ bf16  Vt[32 * HEADD];    //  4 KB staged V tile

    const float scale = 0.125f;         // HEAD^-0.5
    int lane = threadIdx.x;
    int half = lane >> 4, lr = lane & 15;
    int bid = blockIdx.x;
    int qt = bid & 31;                  // query tile (SEQ/16 = 32)
    int hh = (bid >> 5) & (N_HEAD - 1);
    int b  = bid >> 9;

    // --- load the two A (q) fragments once (K = 64 -> 2 k-steps) ----------
    const bf16* qrow = Q + (size_t)(b * SEQ + qt * 16 + lr) * HIDDEN + hh * HEADD + half * 8;
    bf16x16 aq0 = load_frag_rowmajor(qrow);
    bf16x16 aq1 = load_frag_rowmajor(qrow + 32);

    // --- scores ------------------------------------------------------------
    for (int nt = 0; nt < SEQ / 16; ++nt) {
        int key = nt * 16 + lr;
        const bf16* krow = Kb + (size_t)(b * SEQ + key) * HIDDEN + hh * HEADD + half * 8;
        f32x8 acc = {};
        bf16x16 bk0 = load_frag_rowmajor(krow);
        acc = __builtin_amdgcn_wmma_f32_16x16x32_bf16(false, aq0, false, bk0,
                                                      (short)0, acc, false, false);
        bf16x16 bk1 = load_frag_rowmajor(krow + 32);
        acc = __builtin_amdgcn_wmma_f32_16x16x32_bf16(false, aq1, false, bk1,
                                                      (short)0, acc, false, false);
#pragma unroll
        for (int r = 0; r < 8; ++r)
            S[(half * 8 + r) * SEQ + nt * 16 + lr] = acc[r] * scale;
    }
    __syncthreads();

    // --- softmax: lane handles row lr, columns [half*256, half*256+256) ----
    {
        int r = lr;
        int c0 = half * 256;
        float mx = -3.4e38f;
        for (int j = 0; j < 256; ++j) mx = fmaxf(mx, S[r * SEQ + c0 + j]);
        mx = fmaxf(mx, __shfl_xor(mx, 16, 32));
        float sum = 0.f;
        for (int j = 0; j < 256; ++j) {
            float e = expf(S[r * SEQ + c0 + j] - mx);
            S[r * SEQ + c0 + j] = e;
            sum += e;
        }
        sum += __shfl_xor(sum, 16, 32);
        float inv = 1.0f / sum;
        for (int j = 0; j < 256; ++j)
            P[r * SEQ + c0 + j] = (bf16)(S[r * SEQ + c0 + j] * inv);
    }
    __syncthreads();

    // --- att = P @ V --------------------------------------------------------
    f32x8 accv[4];
#pragma unroll
    for (int t = 0; t < 4; ++t) accv[t] = (f32x8){};

    for (int ks = 0; ks < SEQ / 32; ++ks) {
        // stage V rows [ks*32 .. ks*32+31] into LDS: one row per lane
        const bf16x8* vrow = (const bf16x8*)(Vb +
            (size_t)(b * SEQ + ks * 32 + lane) * HIDDEN + hh * HEADD);
        bf16x8* vdst = (bf16x8*)&Vt[lane * HEADD];
#pragma unroll
        for (int t = 0; t < 8; ++t) vdst[t] = vrow[t];
        __syncthreads();

        // A fragment from probability matrix in LDS (row-major, ld = 512)
        bf16x16 ap = load_frag_rowmajor(&P[lr * SEQ + ks * 32 + half * 8]);

#pragma unroll
        for (int nt2 = 0; nt2 < 4; ++nt2) {
            int e = nt2 * 16 + lr;
            bf16x16 bv;
#pragma unroll
            for (int i = 0; i < 16; ++i) {
                int kk = (i < 8) ? (half * 8 + i) : (16 + half * 8 + (i - 8));
                bv[i] = Vt[kk * HEADD + e];
            }
            accv[nt2] = __builtin_amdgcn_wmma_f32_16x16x32_bf16(
                false, ap, false, bv, (short)0, accv[nt2], false, false);
        }
        __syncthreads();
    }

    // --- epilogue: head-concatenated layout [b*512+l][h*64+e] ---------------
#pragma unroll
    for (int nt2 = 0; nt2 < 4; ++nt2) {
#pragma unroll
        for (int r = 0; r < 8; ++r) {
            int l = qt * 16 + half * 8 + r;
            Ab[(size_t)(b * SEQ + l) * HIDDEN + hh * HEADD + nt2 * 16 + lr] =
                (bf16)accv[nt2][r];
        }
    }
}

// ---------------------------------------------------------------------------
// Host-side orchestration (graph-capture safe: only launches on `stream`).
// ---------------------------------------------------------------------------
extern "C" void kernel_launch(void* const* d_in, const int* in_sizes, int n_in,
                              void* d_out, int out_size, void* d_ws, size_t ws_size,
                              hipStream_t stream) {
    (void)in_sizes; (void)n_in; (void)out_size; (void)ws_size;
    const float* x   = (const float*)d_in[0];
    const float* emb = (const float*)d_in[1];
    const float* pos = (const float*)d_in[2];
    const float* Wq  = (const float*)d_in[3];
    const float* Wk  = (const float*)d_in[4];
    const float* Wv  = (const float*)d_in[5];
    const float* Wo  = (const float*)d_in[6];
    const float* bo  = (const float*)d_in[7];
    const float* W1  = (const float*)d_in[8];
    const float* b1  = (const float*)d_in[9];
    const float* W2  = (const float*)d_in[10];
    const float* b2  = (const float*)d_in[11];
    const float* g1  = (const float*)d_in[12];
    const float* g2  = (const float*)d_in[13];

    // workspace carve-up
    char* w = (char*)d_ws;
    float* h   = (float*)w; w += (size_t)MROWS * HIDDEN * 4;   // 32 MB
    bf16* xn   = (bf16*)w;  w += (size_t)MROWS * HIDDEN * 2;   // 16 MB
    bf16* qb   = (bf16*)w;  w += (size_t)MROWS * HIDDEN * 2;
    bf16* kb   = (bf16*)w;  w += (size_t)MROWS * HIDDEN * 2;
    bf16* vb   = (bf16*)w;  w += (size_t)MROWS * HIDDEN * 2;
    bf16* attb = (bf16*)w;  w += (size_t)MROWS * HIDDEN * 2;
    bf16* ffb  = (bf16*)w;  w += (size_t)MROWS * FFWD * 2;     // 32 MB
    bf16* pQ   = (bf16*)w;  w += (size_t)HIDDEN * HIDDEN * 2;  // 2 MB each
    bf16* pK   = (bf16*)w;  w += (size_t)HIDDEN * HIDDEN * 2;
    bf16* pV   = (bf16*)w;  w += (size_t)HIDDEN * HIDDEN * 2;
    bf16* pO   = (bf16*)w;  w += (size_t)HIDDEN * HIDDEN * 2;
    bf16* p1   = (bf16*)w;  w += (size_t)HIDDEN * FFWD * 2;    // 4 MB
    bf16* p2   = (bf16*)w;  w += (size_t)FFWD * HIDDEN * 2;    // 4 MB

    embed_kernel<<<MROWS, 256, 0, stream>>>(x, emb, pos, h);

    // waves = (M/64) * (N/32), 8 waves (256 threads) per block
    const int gQKV = (MROWS / 64) * (HIDDEN / 32) / 8;  // 512 blocks
    const int gFF1 = (MROWS / 64) * (FFWD / 32) / 8;    // 1024 blocks

    for (int L = 0; L < N_LAYER; ++L) {
        // pack this layer's weights into WMMA B-fragment order (bf16)
        pack_weight<<<(HIDDEN * HIDDEN) / 256, 256, 0, stream>>>(
            Wq + (size_t)L * N_HEAD * HIDDEN * HEADD, pQ, HIDDEN, HIDDEN, 1);
        pack_weight<<<(HIDDEN * HIDDEN) / 256, 256, 0, stream>>>(
            Wk + (size_t)L * N_HEAD * HIDDEN * HEADD, pK, HIDDEN, HIDDEN, 1);
        pack_weight<<<(HIDDEN * HIDDEN) / 256, 256, 0, stream>>>(
            Wv + (size_t)L * N_HEAD * HIDDEN * HEADD, pV, HIDDEN, HIDDEN, 1);
        pack_weight<<<(HIDDEN * HIDDEN) / 256, 256, 0, stream>>>(
            Wo + (size_t)L * HIDDEN * HIDDEN, pO, HIDDEN, HIDDEN, 0);
        pack_weight<<<(HIDDEN * FFWD) / 256, 256, 0, stream>>>(
            W1 + (size_t)L * HIDDEN * FFWD, p1, HIDDEN, FFWD, 0);
        pack_weight<<<(FFWD * HIDDEN) / 256, 256, 0, stream>>>(
            W2 + (size_t)L * FFWD * HIDDEN, p2, FFWD, HIDDEN, 0);

        // attention block
        rmsnorm_kernel<<<MROWS, 256, 0, stream>>>(h, g1 + (size_t)L * HIDDEN, xn);
        gemm_bf16_wmma<0><<<gQKV, 256, 0, stream>>>(xn, pQ, nullptr, nullptr,
                                                    qb, nullptr, MROWS, HIDDEN, HIDDEN);
        gemm_bf16_wmma<0><<<gQKV, 256, 0, stream>>>(xn, pK, nullptr, nullptr,
                                                    kb, nullptr, MROWS, HIDDEN, HIDDEN);
        gemm_bf16_wmma<0><<<gQKV, 256, 0, stream>>>(xn, pV, nullptr, nullptr,
                                                    vb, nullptr, MROWS, HIDDEN, HIDDEN);
        attention_kernel<<<BATCH * N_HEAD * (SEQ / 16), 32, 0, stream>>>(qb, kb, vb, attb);
        gemm_bf16_wmma<1><<<gQKV, 256, 0, stream>>>(attb, pO, bo + (size_t)L * HIDDEN,
                                                    h, nullptr, h, MROWS, HIDDEN, HIDDEN);

        // feed-forward block
        rmsnorm_kernel<<<MROWS, 256, 0, stream>>>(h, g2 + (size_t)L * HIDDEN, xn);
        gemm_bf16_wmma<2><<<gFF1, 256, 0, stream>>>(xn, p1, b1 + (size_t)L * FFWD,
                                                    nullptr, ffb, nullptr, MROWS, FFWD, HIDDEN);
        float* outp = (L == N_LAYER - 1) ? (float*)d_out : h;
        gemm_bf16_wmma<1><<<gQKV, 256, 0, stream>>>(ffb, p2, b2 + (size_t)L * HIDDEN,
                                                    h, nullptr, outp, MROWS, HIDDEN, FFWD);
    }
}
